// ViLBert3DMF_20933670601445
// MI455X (gfx1250) — compile-verified, matrix operand order checked by reading, not run on hardware
//
#include <hip/hip_runtime.h>
#include <hip/hip_bf16.h>

typedef __bf16 bf16_t;
typedef __attribute__((ext_vector_type(16))) __bf16 v16bf;
typedef __attribute__((ext_vector_type(8)))  __bf16 v8bf;
typedef __attribute__((ext_vector_type(8)))  float  v8f;

// ---------------- workspace layout (bytes) ----------------
// Region R1 (92,274,688 B) is used twice:
//   phase A: featbf (46,137,344) + prevbf (46,137,344)     [bf16 concat feats]
//   phase B: ximg   (67,108,864) + xpc   (25,165,824)      [bf16 gathered inputs]
static constexpr size_t OFF_FEAT  = 0;
static constexpr size_t OFF_PREV  = 46137344;
static constexpr size_t OFF_XIMG  = 0;
static constexpr size_t OFF_XPC   = 67108864;
static constexpr size_t OFF_SIM   = 92274688;   //  8,388,608  f32 [32][256][256]
static constexpr size_t OFF_IW1T  = 100663296;  // 16,777,216  bf16 [2048][4096]
static constexpr size_t OFF_IW2T  = 117440512;  //  8,388,608  bf16 [2048][2048]
static constexpr size_t OFF_IW3T  = 125829120;  //  4,194,304  bf16 [1024][2048]
static constexpr size_t OFF_PW1T  = 130023424;  //  2,359,296  bf16 [768][1536]
static constexpr size_t OFF_PW2T  = 132382720;  //  1,179,648  bf16 [768][768]
static constexpr size_t OFF_PW3T  = 133562368;  //  1,572,864  bf16 [1024][768]
static constexpr size_t OFF_FW1T  = 135135232;  //  4,194,304  bf16 [1024][2048]
static constexpr size_t OFF_FW2T  = 139329536;  //  2,097,152  bf16 [1024][1024]
static constexpr size_t OFF_H1    = 141426688;  // 33,554,432  bf16 scratch (ping)
static constexpr size_t OFF_H2    = 174981120;  // 33,554,432  bf16 scratch (pong)
static constexpr size_t OFF_XF    = 208535552;  // 33,554,432  bf16 [8192][2048] LN outputs
static constexpr size_t OFF_NORMF = 242089984;  //     32,768  f32 [8192]
static constexpr size_t OFF_NORMP = 242122752;  //     32,768  f32 [8192]
static constexpr size_t OFF_MIDX  = 242155520;  //     32,768  i32 [8192]

// ---------------- fragment helpers ----------------
// 16-bit A/B layout (ISA 7.12.2): lane = (m = lane&15, half = lane>>4).
// VGPR v (0..7) holds K = (v&3)*2 + (v>=4 ? 16 : 0) + half*8, +0/+1.
// => per lane: chunks [kb+half*8 .. +7] and [kb+half*8+16 .. +7] (two 16B loads)
__device__ __forceinline__ v16bf cat8(v8bf c0, v8bf c1) {
  return __builtin_shufflevector(c0, c1, 0, 1, 2, 3, 4, 5, 6, 7,
                                         8, 9, 10, 11, 12, 13, 14, 15);
}
__device__ __forceinline__ v16bf load_frag(const bf16_t* __restrict__ row,
                                           int kb, int half) {
  const v8bf c0 = *(const v8bf*)(row + kb + half * 8);
  const v8bf c1 = *(const v8bf*)(row + kb + half * 8 + 16);
  return cat8(c0, c1);
}

// ---------------- shared GEMM core ----------------
// out tile per wave: 16 rows x 64 cols; block = 8 waves -> 128 x 64 tile.
// B tile (64 cols x 32 k, bf16, 4 KB) is cooperatively staged in LDS and
// double-buffered: loads for step i+1 are issued before the 4 WMMAs of
// step i, and ds_store'd into the alternate buffer after the math.
// All 8 waves read identical B fragments from LDS (8x less L2 traffic).
__device__ __forceinline__ void gemm_core(const bf16_t* __restrict__ A,
                                          const bf16_t* __restrict__ BT,
                                          int K, int tileM, int tileN,
                                          v8bf (*ldsB)[64][4],   // [2][64][4]
                                          v8f acc[4]) {
  const int t    = threadIdx.x;
  const int lane = t & 31;
  const int m    = lane & 15, half = lane >> 4;
  const int srow = t >> 2, schunk = t & 3;         // staging: 64 rows x 4 chunks
  const bf16_t* arow   = A  + (size_t)(tileM + m) * K;
  const bf16_t* bstage = BT + (size_t)(tileN + srow) * K + schunk * 8;

  const int nsteps = K >> 5;
  // prologue: stage step 0 into buffer 0, prefetch A fragment 0
  ldsB[0][srow][schunk] = *(const v8bf*)(bstage);
  v16bf aCur = load_frag(arow, 0, half);
  __syncthreads();

  for (int i = 0; i < nsteps; ++i) {
    const int buf = i & 1;
    const bool has_next = (i + 1) < nsteps;
    v8bf snxt;
    v16bf aNxt;
    if (has_next) {                                // issue next-step loads early
      const int kn = (i + 1) << 5;
      snxt = *(const v8bf*)(bstage + kn);
      aNxt = load_frag(arow, kn, half);
    }
#pragma unroll
    for (int s = 0; s < 4; ++s) {                  // math for step i (hides loads)
      const v8bf c0 = ldsB[buf][s * 16 + m][half];
      const v8bf c1 = ldsB[buf][s * 16 + m][half + 2];
      acc[s] = __builtin_amdgcn_wmma_f32_16x16x32_bf16(
          false, aCur, false, cat8(c0, c1), (short)0, acc[s], false, false);
    }
    if (has_next) {
      ldsB[buf ^ 1][srow][schunk] = snxt;
      aCur = aNxt;
    }
    __syncthreads();
  }
}

// ---------------- weight convert + transpose: W[K][N] f32 -> WT[N][K] bf16 ----
__global__ __launch_bounds__(256)
void wtrans_kernel(const float* __restrict__ w, bf16_t* __restrict__ wt,
                   int K, int N) {
  size_t i = (size_t)blockIdx.x * blockDim.x + threadIdx.x;
  if (i >= (size_t)K * N) return;
  int k = (int)(i / N), n = (int)(i % N);
  wt[(size_t)n * K + k] = (bf16_t)w[i];
}

// -------- concat(img,pc) -> bf16 row + L2 norm (clamped at EPS) --------
__global__ __launch_bounds__(256)
void prep_features_kernel(const float* __restrict__ img,
                          const float* __restrict__ pc,
                          bf16_t* __restrict__ outbf,
                          float* __restrict__ norms) {
  const int row = blockIdx.x;                         // B*N rows
  const float* ip = img + (size_t)row * 2048;
  const float* pp = pc  + (size_t)row * 768;
  bf16_t* orow = outbf + (size_t)row * 2816;
  float ss = 0.f;
  for (int j = threadIdx.x; j < 2816; j += 256) {
    float v = (j < 2048) ? ip[j] : pp[j - 2048];
    ss += v * v;
    orow[j] = (bf16_t)v;
  }
  __shared__ float red[256];
  red[threadIdx.x] = ss; __syncthreads();
  for (int s = 128; s > 0; s >>= 1) {
    if (threadIdx.x < s) red[threadIdx.x] += red[threadIdx.x + s];
    __syncthreads();
  }
  if (threadIdx.x == 0) norms[row] = fmaxf(sqrtf(red[0]), 1e-8f);
}

// -------- batched cosine-sim GEMM: sim[b] = feat[b] (256xK) * prev[b]^T ------
__global__ __launch_bounds__(256)
void sim_gemm_kernel(const bf16_t* __restrict__ featbf,
                     const bf16_t* __restrict__ prevbf,
                     const float* __restrict__ normf,
                     const float* __restrict__ normp,
                     float* __restrict__ sim) {
  __shared__ v8bf ldsB[2][64][4];
  const int b    = blockIdx.z;
  const int lane = threadIdx.x & 31;
  const int wave = threadIdx.x >> 5;
  const int m    = lane & 15, half = lane >> 4;
  const int tileM = blockIdx.y * 128 + wave * 16;
  const int tileN = blockIdx.x * 64;
  const bf16_t* A  = featbf + (size_t)b * 256 * 2816;
  const bf16_t* BT = prevbf + (size_t)b * 256 * 2816;

  v8f acc[4] = {};
  gemm_core(A, BT, 2816, tileM, tileN, ldsB, acc);

#pragma unroll
  for (int s = 0; s < 4; ++s) {
    const int col = tileN + s * 16 + m;
    const float np = normp[b * 256 + col];
#pragma unroll
    for (int r = 0; r < 8; ++r) {
      const int row = tileM + half * 8 + r;
      const float nf = normf[b * 256 + row];
      sim[((size_t)b * 256 + row) * 256 + col] = acc[s][r] / (nf * np);
    }
  }
}

// -------- greedy match, one block per batch (sequential over N rows) --------
__global__ __launch_bounds__(256)
void greedy_kernel(const float* __restrict__ sim,
                   const unsigned char* __restrict__ mask,
                   const float* __restrict__ prev_spatial,
                   int* __restrict__ match_idx,
                   float* __restrict__ out_spatial) {
  const int b = blockIdx.x;
  const int t = threadIdx.x;
  __shared__ int   visited[256];
  __shared__ float s_score[256];
  __shared__ int   s_idx[256];
  __shared__ int   s_found, s_sel;
  visited[t] = 0;
  __syncthreads();
  const float* simb = sim + (size_t)b * 65536;
  const unsigned char* maskb = mask + (size_t)b * 65536;
  for (int i = 0; i < 256; ++i) {
    const float sv = simb[i * 256 + t];
    const bool valid = (maskb[i * 256 + t] != 0) && (!visited[t]) && (sv >= 0.5f);
    s_score[t] = valid ? sv : -__builtin_inff();
    s_idx[t] = t;
    __syncthreads();
    for (int s = 128; s > 0; s >>= 1) {
      if (t < s) {
        const float a = s_score[t], c = s_score[t + s];
        if (c > a || (c == a && s_idx[t + s] < s_idx[t])) {  // argmax, first index
          s_score[t] = c; s_idx[t] = s_idx[t + s];
        }
      }
      __syncthreads();
    }
    if (t == 0) {
      const int found = (s_score[0] > -__builtin_inff()) ? 1 : 0;
      s_found = found; s_sel = s_idx[0];
      if (found) visited[s_idx[0]] = 1;
      match_idx[b * 256 + i] = found ? s_idx[0] : -1;
    }
    __syncthreads();
    if (t < 8) {
      float v = 0.f;
      if (s_found) v = prev_spatial[((size_t)b * 256 + s_sel) * 8 + t];
      out_spatial[((size_t)b * 256 + i) * 8 + t] = v;
    }
    __syncthreads();
  }
}

// -------- build x = bf16(concat(cur_row, gathered_prev_row or 0)) --------
__global__ __launch_bounds__(256)
void build_x_kernel(const float* __restrict__ cur,
                    const float* __restrict__ prev,
                    const int* __restrict__ match_idx,
                    bf16_t* __restrict__ out, int D) {
  const int row = blockIdx.x;          // B*N
  const int idx = match_idx[row];
  const int b = row >> 8;
  const float* c = cur + (size_t)row * D;
  const float* p = prev + ((size_t)b * 256 + (idx >= 0 ? idx : 0)) * D;
  bf16_t* o = out + (size_t)row * (2 * D);
  for (int j = threadIdx.x; j < D; j += 256) {
    o[j] = (bf16_t)c[j];
    o[D + j] = (bf16_t)((idx >= 0) ? p[j] : 0.f);
  }
}

// -------- dense GEMM: out[M x N] = A[M x K] * BT[N x K]^T + bias (opt relu) --
__global__ __launch_bounds__(256)
void mlp_gemm_kernel(const bf16_t* __restrict__ A,
                     const bf16_t* __restrict__ BT,
                     const float* __restrict__ bias,
                     bf16_t* __restrict__ out,
                     int K, int N, int relu) {
  __shared__ v8bf ldsB[2][64][4];
  const int lane = threadIdx.x & 31;
  const int wave = threadIdx.x >> 5;
  const int m    = lane & 15, half = lane >> 4;
  const int tileM = blockIdx.y * 128 + wave * 16;
  const int tileN = blockIdx.x * 64;

  v8f acc[4] = {};
  gemm_core(A, BT, K, tileM, tileN, ldsB, acc);

#pragma unroll
  for (int s = 0; s < 4; ++s) {
    const int col = tileN + s * 16 + m;
    const float bv = bias[col];
#pragma unroll
    for (int r = 0; r < 8; ++r) {
      const int row = tileM + half * 8 + r;
      float v = acc[s][r] + bv;
      if (relu) v = fmaxf(v, 0.f);
      out[(size_t)row * N + col] = (bf16_t)v;
    }
  }
}

// -------- layernorm over D=1024; writes bf16 (strided) or f32 --------
__global__ __launch_bounds__(256)
void layernorm_kernel(const bf16_t* __restrict__ x,
                      const float* __restrict__ g,
                      const float* __restrict__ beta,
                      bf16_t* __restrict__ out_bf, int out_stride, int out_off,
                      float* __restrict__ out_f) {
  const int row = blockIdx.x;                  // 8192 rows
  const int t = threadIdx.x;
  const bf16_t* xr = x + (size_t)row * 1024;
  float vals[4], s = 0.f, ss = 0.f;
#pragma unroll
  for (int j = 0; j < 4; ++j) {
    float v = (float)xr[t + j * 256];
    vals[j] = v; s += v; ss += v * v;
  }
  __shared__ float r1[256], r2[256];
  r1[t] = s; r2[t] = ss; __syncthreads();
  for (int st = 128; st > 0; st >>= 1) {
    if (t < st) { r1[t] += r1[t + st]; r2[t] += r2[t + st]; }
    __syncthreads();
  }
  const float mean = r1[0] * (1.f / 1024.f);
  const float var  = r2[0] * (1.f / 1024.f) - mean * mean;
  const float rstd = rsqrtf(var + 1e-5f);
#pragma unroll
  for (int j = 0; j < 4; ++j) {
    const int col = t + j * 256;
    const float y = (vals[j] - mean) * rstd * g[col] + beta[col];
    if (out_bf) out_bf[(size_t)row * out_stride + out_off + col] = (bf16_t)y;
    else        out_f[(size_t)row * 1024 + col] = y;
  }
}

// ---------------------------------------------------------------------------
extern "C" void kernel_launch(void* const* d_in, const int* in_sizes, int n_in,
                              void* d_out, int out_size, void* d_ws, size_t ws_size,
                              hipStream_t stream) {
  (void)in_sizes; (void)n_in; (void)out_size; (void)ws_size;
  const float* imgf  = (const float*)d_in[0];
  const float* pcf   = (const float*)d_in[1];
  const float* pimgf = (const float*)d_in[2];
  const float* ppcf  = (const float*)d_in[3];
  const float* pspat = (const float*)d_in[4];
  const unsigned char* mask = (const unsigned char*)d_in[5];
  const float* iw1 = (const float*)d_in[6];  const float* ib1 = (const float*)d_in[7];
  const float* iw2 = (const float*)d_in[8];  const float* ib2 = (const float*)d_in[9];
  const float* iw3 = (const float*)d_in[10]; const float* ib3 = (const float*)d_in[11];
  const float* ig  = (const float*)d_in[12]; const float* ibeta = (const float*)d_in[13];
  const float* pw1 = (const float*)d_in[14]; const float* pb1 = (const float*)d_in[15];
  const float* pw2 = (const float*)d_in[16]; const float* pb2 = (const float*)d_in[17];
  const float* pw3 = (const float*)d_in[18]; const float* pb3 = (const float*)d_in[19];
  const float* pg  = (const float*)d_in[20]; const float* pbeta = (const float*)d_in[21];
  const float* fw1 = (const float*)d_in[22]; const float* fb1 = (const float*)d_in[23];
  const float* fw2 = (const float*)d_in[24]; const float* fb2 = (const float*)d_in[25];
  const float* fg  = (const float*)d_in[26]; const float* fbeta = (const float*)d_in[27];

  char* ws = (char*)d_ws;
  bf16_t* featbf = (bf16_t*)(ws + OFF_FEAT);
  bf16_t* prevbf = (bf16_t*)(ws + OFF_PREV);
  bf16_t* ximg   = (bf16_t*)(ws + OFF_XIMG);
  bf16_t* xpc    = (bf16_t*)(ws + OFF_XPC);
  float*  sim    = (float*)(ws + OFF_SIM);
  bf16_t* iw1t = (bf16_t*)(ws + OFF_IW1T);
  bf16_t* iw2t = (bf16_t*)(ws + OFF_IW2T);
  bf16_t* iw3t = (bf16_t*)(ws + OFF_IW3T);
  bf16_t* pw1t = (bf16_t*)(ws + OFF_PW1T);
  bf16_t* pw2t = (bf16_t*)(ws + OFF_PW2T);
  bf16_t* pw3t = (bf16_t*)(ws + OFF_PW3T);
  bf16_t* fw1t = (bf16_t*)(ws + OFF_FW1T);
  bf16_t* fw2t = (bf16_t*)(ws + OFF_FW2T);
  bf16_t* h1 = (bf16_t*)(ws + OFF_H1);
  bf16_t* h2 = (bf16_t*)(ws + OFF_H2);
  bf16_t* xf = (bf16_t*)(ws + OFF_XF);
  float*  normf = (float*)(ws + OFF_NORMF);
  float*  normp = (float*)(ws + OFF_NORMP);
  int*    midx  = (int*)(ws + OFF_MIDX);

  float* out_vis  = (float*)d_out;              // [32*256*1024]
  float* out_spat = out_vis + 8388608;          // [32*256*8]

  auto wt = [&](const float* w, bf16_t* o, int K, int N) {
    size_t n = (size_t)K * N;
    wtrans_kernel<<<dim3((unsigned)((n + 255) / 256)), 256, 0, stream>>>(w, o, K, N);
  };
  // 1. weights: f32 [K][N] -> bf16 [N][K]
  wt(iw1, iw1t, 4096, 2048); wt(iw2, iw2t, 2048, 2048); wt(iw3, iw3t, 2048, 1024);
  wt(pw1, pw1t, 1536, 768);  wt(pw2, pw2t, 768, 768);   wt(pw3, pw3t, 768, 1024);
  wt(fw1, fw1t, 2048, 1024); wt(fw2, fw2t, 1024, 1024);

  // 2. bf16 concat features + row norms
  prep_features_kernel<<<8192, 256, 0, stream>>>(imgf, pcf, featbf, normf);
  prep_features_kernel<<<8192, 256, 0, stream>>>(pimgf, ppcf, prevbf, normp);

  // 3. batched similarity GEMM (bf16 WMMA, normalized in epilogue)
  sim_gemm_kernel<<<dim3(4, 2, 32), 256, 0, stream>>>(featbf, prevbf, normf, normp, sim);

  // 4. greedy matching: match indices + new_prev_spatial (direct to d_out)
  greedy_kernel<<<32, 256, 0, stream>>>(sim, mask, pspat, midx, out_spat);

  // 5. gather + concat inputs (reuses feat region — features are dead now)
  build_x_kernel<<<8192, 256, 0, stream>>>(imgf, pimgf, midx, ximg, 2048);
  build_x_kernel<<<8192, 256, 0, stream>>>(pcf, ppcf, midx, xpc, 768);

  // 6. image branch: relu(x@w1+b1)@w2+b2 @w3+b3 -> LN -> xf[:, :1024]
  mlp_gemm_kernel<<<dim3(2048 / 64, 64), 256, 0, stream>>>(ximg, iw1t, ib1, h1, 4096, 2048, 1);
  mlp_gemm_kernel<<<dim3(2048 / 64, 64), 256, 0, stream>>>(h1, iw2t, ib2, h2, 2048, 2048, 0);
  mlp_gemm_kernel<<<dim3(1024 / 64, 64), 256, 0, stream>>>(h2, iw3t, ib3, h1, 2048, 1024, 0);
  layernorm_kernel<<<8192, 256, 0, stream>>>(h1, ig, ibeta, xf, 2048, 0, nullptr);

  // 7. point-cloud branch -> LN -> xf[:, 1024:]
  mlp_gemm_kernel<<<dim3(768 / 64, 64), 256, 0, stream>>>(xpc, pw1t, pb1, h2, 1536, 768, 1);
  mlp_gemm_kernel<<<dim3(768 / 64, 64), 256, 0, stream>>>(h2, pw2t, pb2, h1, 768, 768, 0);
  mlp_gemm_kernel<<<dim3(1024 / 64, 64), 256, 0, stream>>>(h1, pw3t, pb3, h2, 768, 1024, 0);
  layernorm_kernel<<<8192, 256, 0, stream>>>(h2, pg, pbeta, xf, 2048, 1024, nullptr);

  // 8. fusion: relu(xf@fw1+fb1)@fw2+fb2 -> LN -> d_out (f32)
  mlp_gemm_kernel<<<dim3(1024 / 64, 64), 256, 0, stream>>>(xf, fw1t, fb1, h1, 2048, 1024, 1);
  mlp_gemm_kernel<<<dim3(1024 / 64, 64), 256, 0, stream>>>(h1, fw2t, fb2, h2, 1024, 1024, 0);
  layernorm_kernel<<<8192, 256, 0, stream>>>(h2, fg, fbeta, nullptr, 0, 0, out_vis);
}
